// MambaLayer_15616501088732
// MI455X (gfx1250) — matrix operand order repaired and checked by reading, not compiled, not run on hardware
//
#include <hip/hip_runtime.h>
#include <hip/hip_bf16.h>
#include <math.h>

// ---------------------------------------------------------------------------
// Mamba layer forward for MI455X (gfx1250, wave32, WMMA + TDM).
// Pipeline: LN -> in_proj (WMMA) -> conv+SiLU -> x_proj (WMMA) ->
//           dt_proj+softplus (WMMA) -> selective scan (+SiLU gate fused) ->
//           out_proj + residual (WMMA).
// GEMM tiles are staged into LDS by the Tensor Data Mover (double-buffered,
// TENSORcnt-tracked) when the builtin is available; manual staging otherwise.
// ---------------------------------------------------------------------------

#define N_EMBD 768
#define D_STATE 16
#define D_CONV 4
#define D_INNER 1536
#define DT_RANK 48
#define BB 8
#define LL 2048
#define MROWS (BB * LL)          // 16384 token rows

typedef __attribute__((ext_vector_type(16))) __bf16 v16bf;
typedef __attribute__((ext_vector_type(8)))  float  v8f;
typedef __attribute__((ext_vector_type(4)))  unsigned int u32x4;
typedef __attribute__((ext_vector_type(8)))  int i32x8;
typedef __attribute__((ext_vector_type(4)))  int i32x4;

union FragBf { v16bf v; uint4 q[2]; };

#if defined(__has_builtin)
#if __has_builtin(__builtin_amdgcn_tensor_load_to_lds) && __has_builtin(__builtin_amdgcn_s_wait_tensorcnt)
#define HAVE_TDM 1
#endif
#endif
#ifndef HAVE_TDM
#define HAVE_TDM 0
#endif

#if HAVE_TDM
// 2-D TDM load: tile_rows x tile_k bf16 elements from row-major global
// (row stride = stride_k elements) into LDS at lds_byte_off, with LDS padding
// of 4 dwords every 16 dwords (=> 80-byte padded row stride, matches [..][40]
// bf16 LDS tiles). rem_k / rem_rows express the remaining valid tensor extent
// from the tile origin so hardware OOB handling zero-fills ragged edges.
// Toolchain uses the 6-arg builtin form:
//   (u32x4 g0, i32x8 g1, i32x4 g2, i32x4 g3, i32x8 g4, i32 cpol)
__device__ inline void tdm_load_2d(unsigned int lds_byte_off, const __bf16* g,
                                   unsigned int rem_k, unsigned int rem_rows,
                                   unsigned int tile_k, unsigned int tile_rows,
                                   unsigned int stride_k) {
    unsigned long long ga = (unsigned long long)(uintptr_t)g;
    u32x4 g0;
    g0[0] = 1u;                                           // count=1, user D#
    g0[1] = lds_byte_off;                                 // lds_addr
    g0[2] = (unsigned int)(ga & 0xFFFFFFFFu);             // global_addr[31:0]
    g0[3] = (unsigned int)((ga >> 32) & 0x1FFFFFFu)       // global_addr[56:32]
          | (2u << 30);                                   // type=2 (image)
    i32x8 g1;
    // word0: data_size=1 (2B) @bits17:16, pad_enable @bit20,
    //        pad_interval=3 (16 dwords) @bits24:22, pad_amount=3 (4 dwords) @bits31:25
    g1[0] = (int)((1u << 16) | (1u << 20) | (3u << 22) | (3u << 25));
    g1[1] = (int)((rem_k & 0xFFFFu) << 16);               // tensor_dim0[15:0]
    g1[2] = (int)((rem_k >> 16) | ((rem_rows & 0xFFFFu) << 16)); // dim0 hi | dim1 lo
    g1[3] = (int)((rem_rows >> 16) | (tile_k << 16));     // dim1 hi | tile_dim0
    g1[4] = (int)(tile_rows & 0xFFFFu);                   // tile_dim1 (tile_dim2=0)
    g1[5] = (int)stride_k;                                // tensor_dim0_stride[31:0]
    g1[6] = 0;                                            // stride hi | dim1_stride lo
    g1[7] = 0;
    i32x4 z4; z4[0] = 0; z4[1] = 0; z4[2] = 0; z4[3] = 0;
    i32x8 z8; z8[0] = 0; z8[1] = 0; z8[2] = 0; z8[3] = 0;
    z8[4] = 0; z8[5] = 0; z8[6] = 0; z8[7] = 0;
    __builtin_amdgcn_tensor_load_to_lds(g0, g1, z4, z4, z8, 0);
}
#endif

// ----------------------------- weight prep --------------------------------
// fp32 (K x N) row-major -> bf16 transposed (N x K) row-major so GEMM B tiles
// have K contiguous (matches WMMA B-fragment layout).
__global__ void wconv_t_kernel(const float* __restrict__ src, __bf16* __restrict__ dst,
                               int K, int N) {
    size_t idx = (size_t)blockIdx.x * blockDim.x + threadIdx.x;
    size_t total = (size_t)K * N;
    if (idx >= total) return;
    int k = (int)(idx / N);
    int n = (int)(idx % N);
    dst[(size_t)n * K + k] = (__bf16)src[idx];
}

// ------------------------------ LayerNorm ---------------------------------
__global__ void ln_kernel(const float* __restrict__ x, const float* __restrict__ g,
                          const float* __restrict__ b, __bf16* __restrict__ xn) {
    const int row = blockIdx.x;
    const float* xr = x + (size_t)row * N_EMBD;
    __shared__ float red[256];
    const int tid = threadIdx.x;

    float s = 0.f;
    for (int i = tid; i < N_EMBD; i += 256) s += xr[i];
    red[tid] = s; __syncthreads();
    for (int off = 128; off > 0; off >>= 1) {
        if (tid < off) red[tid] += red[tid + off];
        __syncthreads();
    }
    float mu = red[0] * (1.f / N_EMBD);
    __syncthreads();

    float v = 0.f;
    for (int i = tid; i < N_EMBD; i += 256) { float d = xr[i] - mu; v += d * d; }
    red[tid] = v; __syncthreads();
    for (int off = 128; off > 0; off >>= 1) {
        if (tid < off) red[tid] += red[tid + off];
        __syncthreads();
    }
    float rstd = rsqrtf(red[0] * (1.f / N_EMBD) + 1e-5f);

    for (int i = tid; i < N_EMBD; i += 256) {
        float y = (xr[i] - mu) * rstd * g[i] + b[i];
        xn[(size_t)row * N_EMBD + i] = (__bf16)y;
    }
}

// ------------------------------ WMMA GEMM ---------------------------------
// C[M,N] = A[M,K](bf16) * Bt[N,K](bf16)^T   (fp32 accumulate)
// mode 0: plain -> fp32 Cf or bf16 Cb;  mode 1: +bias, softplus;
// mode 2: +resid[gm*N+gn].
// Block tile 128x64, 8 waves (4x2), each wave 32x32 = 2x2 WMMA frags, BK=32.
__global__ __launch_bounds__(256)
void gemm_wmma_bf16(const __bf16* __restrict__ A, const __bf16* __restrict__ Bt,
                    float* __restrict__ Cf, __bf16* __restrict__ Cb,
                    int M, int N, int K, int mode,
                    const float* __restrict__ bias, const float* __restrict__ resid) {
    __shared__ __bf16 As[2][128][40];   // 128 x 32 tile, row padded to 40 (80 B)
    __shared__ __bf16 Bs[2][64][40];    // 64 x 32 tile (N-major, K contiguous)

    const int tid  = threadIdx.x;
    const int lane = tid & 31;
    const int w    = tid >> 5;       // 8 waves
    const int wm   = w & 3;          // wave grid 4 (M) x 2 (N)
    const int wn   = w >> 2;
    const int m0   = blockIdx.y * 128;
    const int n0   = blockIdx.x * 64;
    const int half = lane >> 4;
    const int lr   = lane & 15;

    v8f acc[2][2] = {};
    int buf = 0;

#if HAVE_TDM
    // -------- TDM double-buffered pipeline --------
    const bool lead = (tid < 32);    // wave 0 issues DMA (EXEC-independent)
    const unsigned int aOff[2] = { (unsigned int)(uintptr_t)&As[0][0][0],
                                   (unsigned int)(uintptr_t)&As[1][0][0] };
    const unsigned int bOff[2] = { (unsigned int)(uintptr_t)&Bs[0][0][0],
                                   (unsigned int)(uintptr_t)&Bs[1][0][0] };
    const unsigned int remRowsA = (unsigned int)(M - m0);
    const unsigned int remRowsB = (unsigned int)(N - n0);
    const int nIter = (K + 31) / 32;

    if (lead) {
        tdm_load_2d(aOff[0], A + (size_t)m0 * K, (unsigned int)K, remRowsA, 32u, 128u, (unsigned int)K);
        tdm_load_2d(bOff[0], Bt + (size_t)n0 * K, (unsigned int)K, remRowsB, 32u, 64u, (unsigned int)K);
    }
    for (int it = 0; it < nIter; ++it) {
        const int k1 = (it + 1) * 32;
        if (lead) {
            if (k1 < K) {
                tdm_load_2d(aOff[buf ^ 1], A + (size_t)m0 * K + k1,
                            (unsigned int)(K - k1), remRowsA, 32u, 128u, (unsigned int)K);
                tdm_load_2d(bOff[buf ^ 1], Bt + (size_t)n0 * K + k1,
                            (unsigned int)(K - k1), remRowsB, 32u, 64u, (unsigned int)K);
                __builtin_amdgcn_s_wait_tensorcnt(2);   // current tile landed
            } else {
                __builtin_amdgcn_s_wait_tensorcnt(0);
            }
        }
        __syncthreads();

        FragBf a[2], bfr[2];
        #pragma unroll
        for (int i = 0; i < 2; ++i) {
            int r = wm * 32 + i * 16 + lr;
            a[i].q[0] = *(const uint4*)(&As[buf][r][half * 8]);
            a[i].q[1] = *(const uint4*)(&As[buf][r][16 + half * 8]);
        }
        #pragma unroll
        for (int j = 0; j < 2; ++j) {
            int r = wn * 32 + j * 16 + lr;
            bfr[j].q[0] = *(const uint4*)(&Bs[buf][r][half * 8]);
            bfr[j].q[1] = *(const uint4*)(&Bs[buf][r][16 + half * 8]);
        }
        #pragma unroll
        for (int i = 0; i < 2; ++i)
            #pragma unroll
            for (int j = 0; j < 2; ++j)
                acc[i][j] = __builtin_amdgcn_wmma_f32_16x16x32_bf16(
                    false, a[i].v, false, bfr[j].v, (short)0, acc[i][j],
                    false, false);
        __syncthreads();
        buf ^= 1;
    }
#else
    // -------- manual staging fallback --------
    for (int k0 = 0; k0 < K; k0 += 32) {
        #pragma unroll
        for (int c = 0; c < 2; ++c) {
            int chunk = tid * 2 + c;
            int row = chunk >> 2;
            int col = (chunk & 3) * 8;
            uint4 val = make_uint4(0u, 0u, 0u, 0u);
            int gm = m0 + row, gk = k0 + col;
            if (gm < M && (gk + 8) <= K)
                val = *(const uint4*)(A + (size_t)gm * K + gk);
            *(uint4*)(&As[0][row][col]) = val;
        }
        {
            int row = tid >> 2;
            int col = (tid & 3) * 8;
            uint4 val = make_uint4(0u, 0u, 0u, 0u);
            int gn = n0 + row, gk = k0 + col;
            if (gn < N && (gk + 8) <= K)
                val = *(const uint4*)(Bt + (size_t)gn * K + gk);
            *(uint4*)(&Bs[0][row][col]) = val;
        }
        __syncthreads();

        FragBf a[2], bfr[2];
        #pragma unroll
        for (int i = 0; i < 2; ++i) {
            int r = wm * 32 + i * 16 + lr;
            a[i].q[0] = *(const uint4*)(&As[0][r][half * 8]);
            a[i].q[1] = *(const uint4*)(&As[0][r][16 + half * 8]);
        }
        #pragma unroll
        for (int j = 0; j < 2; ++j) {
            int r = wn * 32 + j * 16 + lr;
            bfr[j].q[0] = *(const uint4*)(&Bs[0][r][half * 8]);
            bfr[j].q[1] = *(const uint4*)(&Bs[0][r][16 + half * 8]);
        }
        #pragma unroll
        for (int i = 0; i < 2; ++i)
            #pragma unroll
            for (int j = 0; j < 2; ++j)
                acc[i][j] = __builtin_amdgcn_wmma_f32_16x16x32_bf16(
                    false, a[i].v, false, bfr[j].v, (short)0, acc[i][j],
                    false, false);
        __syncthreads();
    }
#endif

    // ---- epilogue: C/D layout -> lane holds col lr, rows 8*half + r
    #pragma unroll
    for (int i = 0; i < 2; ++i) {
        #pragma unroll
        for (int j = 0; j < 2; ++j) {
            int gn = n0 + wn * 32 + j * 16 + lr;
            if (gn >= N) continue;
            int gm_base = m0 + wm * 32 + i * 16 + half * 8;
            #pragma unroll
            for (int r = 0; r < 8; ++r) {
                int gm = gm_base + r;
                if (gm >= M) continue;
                float v = acc[i][j][r];
                if (mode == 1) {
                    v += bias[gn];
                    v = (v > 20.f) ? v : log1pf(__expf(v));   // softplus
                } else if (mode == 2) {
                    v += resid[(size_t)gm * N + gn];
                }
                size_t o = (size_t)gm * N + gn;
                if (Cb) Cb[o] = (__bf16)v;
                else    Cf[o] = v;
            }
        }
    }
}

// ----------------------- depthwise causal conv + SiLU ----------------------
__global__ void conv_silu_kernel(const __bf16* __restrict__ xz, const float* __restrict__ w,
                                 const float* __restrict__ cb,
                                 float* __restrict__ xc_f, __bf16* __restrict__ xc_h) {
    size_t idx = (size_t)blockIdx.x * blockDim.x + threadIdx.x;
    if (idx >= (size_t)MROWS * D_INNER) return;
    int d = (int)(idx % D_INNER);
    size_t bl = idx / D_INNER;          // b*L + l
    int l = (int)(bl % LL);
    size_t brow = bl - l;               // b*L

    float acc = cb[d];
    #pragma unroll
    for (int j = 0; j < D_CONV; ++j) {
        int ll = l - (D_CONV - 1) + j;
        if (ll >= 0)
            acc += w[d * D_CONV + j] * (float)xz[(brow + ll) * (2 * D_INNER) + d];
    }
    float sv = acc / (1.f + __expf(-acc));   // SiLU
    xc_f[idx] = sv;
    xc_h[idx] = (__bf16)sv;
}

// -------------------- extract dt columns of xdbl as bf16 -------------------
__global__ void prep_dt_kernel(const float* __restrict__ xdbl, __bf16* __restrict__ dt_h) {
    size_t idx = (size_t)blockIdx.x * blockDim.x + threadIdx.x;
    if (idx >= (size_t)MROWS * DT_RANK) return;
    size_t row = idx / DT_RANK;
    int col = (int)(idx % DT_RANK);
    dt_h[idx] = (__bf16)xdbl[row * (DT_RANK + 2 * D_STATE) + col];
}

// --------------------------- selective scan -------------------------------
// One thread per (b, d) channel; 16-state recurrence in registers.
// B_t / C_t (32 floats) broadcast via LDS in 8-timestep chunks.
// Fuses y += u*D, gating y *= silu(z), writes bf16 y for out_proj GEMM.
__global__ __launch_bounds__(256)
void scan_kernel(const float* __restrict__ delta, const float* __restrict__ u,
                 const float* __restrict__ xdbl, const __bf16* __restrict__ xz,
                 const float* __restrict__ A_log, const float* __restrict__ Dp,
                 __bf16* __restrict__ y) {
    const int b = blockIdx.x;                          // 0..7
    const int d = blockIdx.y * 256 + threadIdx.x;      // 0..1535

    float Arow[D_STATE];
    #pragma unroll
    for (int n = 0; n < D_STATE; ++n)
        Arow[n] = -__expf(A_log[(size_t)d * D_STATE + n]);
    const float Dd = Dp[d];

    float h[D_STATE];
    #pragma unroll
    for (int n = 0; n < D_STATE; ++n) h[n] = 0.f;

    __shared__ float BC[8][2 * D_STATE];               // 8 timesteps x (B|C)
    const size_t base = (size_t)b * LL;
    const int tt = threadIdx.x >> 5;                   // 0..7
    const int cc = threadIdx.x & 31;                   // 0..31

    for (int t0 = 0; t0 < LL; t0 += 8) {
        __syncthreads();
        BC[tt][cc] = xdbl[(base + t0 + tt) * (DT_RANK + 2 * D_STATE) + DT_RANK + cc];
        __syncthreads();

        #pragma unroll
        for (int ti = 0; ti < 8; ++ti) {
            size_t row = base + t0 + ti;
            float dv = delta[row * D_INNER + d];
            float uv = u[row * D_INNER + d];
            float yv = 0.f;
            #pragma unroll
            for (int n = 0; n < D_STATE; ++n) {
                float dA = __expf(dv * Arow[n]);
                h[n] = dA * h[n] + dv * BC[ti][n] * uv;
                yv += h[n] * BC[ti][D_STATE + n];
            }
            yv += uv * Dd;
            float zf = (float)xz[row * (2 * D_INNER) + D_INNER + d];
            yv *= zf / (1.f + __expf(-zf));            // * silu(z)
            y[row * D_INNER + d] = (__bf16)yv;
        }
    }
}

// ------------------------------- launcher ---------------------------------
extern "C" void kernel_launch(void* const* d_in, const int* in_sizes, int n_in,
                              void* d_out, int out_size, void* d_ws, size_t ws_size,
                              hipStream_t stream) {
    const float* x      = (const float*)d_in[0];
    const float* ln_g   = (const float*)d_in[1];
    const float* ln_b   = (const float*)d_in[2];
    const float* W_in   = (const float*)d_in[3];
    const float* conv_w = (const float*)d_in[4];
    const float* conv_b = (const float*)d_in[5];
    const float* W_x    = (const float*)d_in[6];
    const float* W_dt   = (const float*)d_in[7];
    const float* b_dt   = (const float*)d_in[8];
    const float* A_log  = (const float*)d_in[9];
    const float* Dp     = (const float*)d_in[10];
    const float* W_out  = (const float*)d_in[11];
    float* out = (float*)d_out;

    char* ws = (char*)d_ws;
    size_t off = 0;
    auto alloc = [&](size_t bytes) -> void* {
        void* p = ws + off;
        off += (bytes + 255) & ~(size_t)255;
        return p;
    };

    __bf16* wtin  = (__bf16*)alloc((size_t)N_EMBD * 2 * D_INNER * 2);           // 3072 x 768
    __bf16* wtx   = (__bf16*)alloc((size_t)D_INNER * (DT_RANK + 2 * D_STATE) * 2); // 80 x 1536
    __bf16* wtdt  = (__bf16*)alloc((size_t)DT_RANK * D_INNER * 2);              // 1536 x 48
    __bf16* wtout = (__bf16*)alloc((size_t)D_INNER * N_EMBD * 2);               // 768 x 1536
    __bf16* xn    = (__bf16*)alloc((size_t)MROWS * N_EMBD * 2);
    __bf16* xz_h  = (__bf16*)alloc((size_t)MROWS * 2 * D_INNER * 2);
    float*  xc_f  = (float*) alloc((size_t)MROWS * D_INNER * 4);
    __bf16* xc_h  = (__bf16*)alloc((size_t)MROWS * D_INNER * 2);
    float*  xdbl  = (float*) alloc((size_t)MROWS * (DT_RANK + 2 * D_STATE) * 4);
    __bf16* dt_h  = (__bf16*)alloc((size_t)MROWS * DT_RANK * 2);
    float*  delta = (float*) alloc((size_t)MROWS * D_INNER * 4);
    __bf16* y_h   = (__bf16*)alloc((size_t)MROWS * D_INNER * 2);
    (void)ws_size; (void)in_sizes; (void)n_in; (void)out_size;

    // weight prep (bf16 transposed copies)
    {
        size_t n;
        n = (size_t)N_EMBD * 2 * D_INNER;
        wconv_t_kernel<<<(n + 255) / 256, 256, 0, stream>>>(W_in, wtin, N_EMBD, 2 * D_INNER);
        n = (size_t)D_INNER * (DT_RANK + 2 * D_STATE);
        wconv_t_kernel<<<(n + 255) / 256, 256, 0, stream>>>(W_x, wtx, D_INNER, DT_RANK + 2 * D_STATE);
        n = (size_t)DT_RANK * D_INNER;
        wconv_t_kernel<<<(n + 255) / 256, 256, 0, stream>>>(W_dt, wtdt, DT_RANK, D_INNER);
        n = (size_t)D_INNER * N_EMBD;
        wconv_t_kernel<<<(n + 255) / 256, 256, 0, stream>>>(W_out, wtout, D_INNER, N_EMBD);
    }

    // 1) LayerNorm -> bf16
    ln_kernel<<<MROWS, 256, 0, stream>>>(x, ln_g, ln_b, xn);

    // 2) in_proj: xz = xn @ W_in   (16384 x 3072, bf16 out)
    gemm_wmma_bf16<<<dim3((2 * D_INNER) / 64, MROWS / 128), 256, 0, stream>>>(
        xn, wtin, nullptr, xz_h, MROWS, 2 * D_INNER, N_EMBD, 0, nullptr, nullptr);

    // 3) depthwise causal conv + SiLU
    {
        size_t n = (size_t)MROWS * D_INNER;
        conv_silu_kernel<<<(n + 255) / 256, 256, 0, stream>>>(xz_h, conv_w, conv_b, xc_f, xc_h);
    }

    // 4) x_proj: xdbl = xc @ W_x   (16384 x 80, fp32 out)
    gemm_wmma_bf16<<<dim3((DT_RANK + 2 * D_STATE + 63) / 64, MROWS / 128), 256, 0, stream>>>(
        xc_h, wtx, xdbl, nullptr, MROWS, DT_RANK + 2 * D_STATE, D_INNER, 0, nullptr, nullptr);

    // 5) dt slice -> bf16
    {
        size_t n = (size_t)MROWS * DT_RANK;
        prep_dt_kernel<<<(n + 255) / 256, 256, 0, stream>>>(xdbl, dt_h);
    }

    // 6) dt_proj + bias + softplus: delta (16384 x 1536, fp32)
    gemm_wmma_bf16<<<dim3(D_INNER / 64, MROWS / 128), 256, 0, stream>>>(
        dt_h, wtdt, delta, nullptr, MROWS, D_INNER, DT_RANK, 1, b_dt, nullptr);

    // 7) selective scan + u*D + SiLU(z) gate -> y bf16
    scan_kernel<<<dim3(BB, D_INNER / 256), 256, 0, stream>>>(
        delta, xc_f, xdbl, xz_h, A_log, Dp, y_h);

    // 8) out_proj + residual: out = y @ W_out + x
    gemm_wmma_bf16<<<dim3(N_EMBD / 64, MROWS / 128), 256, 0, stream>>>(
        y_h, wtout, out, nullptr, MROWS, N_EMBD, D_INNER, 2, nullptr, x);
}
